// SpectralConv2DTranspose_75703093559773
// MI455X (gfx1250) — matrix operand order repaired and checked by reading, not compiled, not run on hardware
//
#include <hip/hip_runtime.h>
#include <cstddef>

typedef __attribute__((ext_vector_type(16))) _Float16 v16h;
typedef __attribute__((ext_vector_type(8)))  float    v8f;

union FragU { v16h v; unsigned u[8]; };

__device__ inline v8f zero8() {
  v8f z = {0.f,0.f,0.f,0.f,0.f,0.f,0.f,0.f};
  return z;
}

__device__ inline v8f wmma_f16(v16h a, v16h b, v8f c) {
  // v_wmma_f32_16x16x32_f16  (no A/B neg: NEG[1:0] must be 0 for FP16 per ISA)
  return __builtin_amdgcn_wmma_f32_16x16x32_f16(false, a, false, b, (short)0, c, false, false);
}

// A-operand fragment (16x32 f16) from row-major [M][K] array, element (m0+.., k0+..)
// Layout: lane L: M = m0 + (L&15); hi = L>>4; VGPR g holds K pair
//   k = k0 + hi*16 + 2g (+h)
__device__ inline v16h load_frag_a(const _Float16* p, int m0, int k0, int ldk) {
  const int lane = threadIdx.x & 31;
  const int m = m0 + (lane & 15);
  const int hi = lane >> 4;
  const _Float16* row = p + m * ldk + k0 + hi * 16;
  FragU f;
#pragma unroll
  for (int g = 0; g < 8; ++g) {
    int k = g << 1;
    f.u[g] = *(const unsigned*)(row + k);
  }
  return f.v;
}

// B-operand fragment (32x16 f16) from TRANSPOSED storage BT[N][K] (B[k][n] at BT[n*ldk+k])
// Layout: lane L: N = n0 + (L&15); hi = L>>4; VGPR g holds k = k0 + hi*16 + 2g (+h)
__device__ inline v16h load_frag_bt(const _Float16* p, int n0, int k0, int ldk) {
  const int lane = threadIdx.x & 31;
  const int n = n0 + (lane & 15);
  const int hi = lane >> 4;
  const _Float16* row = p + n * ldk + k0 + hi * 16;
  FragU f;
#pragma unroll
  for (int g = 0; g < 8; ++g) f.u[g] = *(const unsigned*)(row + 2 * g);
  return f.v;
}

// ---------------------------------------------------------------------------
// Problem constants
// ---------------------------------------------------------------------------
#define BB 32
#define NCI 64
#define NCO 64
#define HH 128
#define WW 128
#define MM1 32
#define MM2 32
#define WOUT 65
#define WPAD 80
#define NMODE 2048   // 64 selected kx rows * 32 ky cols

// ---------------------------------------------------------------------------
// Kernel: build f16 DFT basis tables (deterministic; rebuilt each launch)
// ---------------------------------------------------------------------------
__global__ void k_init_basis(_Float16* FwCT, _Float16* FwSTn,
                             _Float16* FhC, _Float16* FhS, _Float16* FhSn,
                             _Float16* IhC, _Float16* IhS, _Float16* IhSn,
                             _Float16* IwCT, _Float16* IwSTn) {
  const int idx = blockIdx.x * blockDim.x + threadIdx.x;
  const float TWO_PI = 6.283185307179586f;
  // Forward W basis (transposed [ky][v]): e^{-i 2pi ky v / 128}
  if (idx < MM2 * WW) {
    int ky = idx >> 7, v = idx & 127;
    float ang = TWO_PI * (float)((ky * v) & 127) * (1.0f / 128.0f);
    FwCT[idx]  = (_Float16)cosf(ang);
    FwSTn[idx] = (_Float16)(-sinf(ang));
  }
  // Forward H basis [kxsel][u]: e^{-i 2pi kxa u / 128}, kxa = kxsel or kxsel+64
  if (idx < 64 * HH) {
    int kxs = idx >> 7, u = idx & 127;
    int kxa = (kxs < 32) ? kxs : kxs + 64;
    float ang = TWO_PI * (float)((kxa * u) & 127) * (1.0f / 128.0f);
    float c = cosf(ang), s = sinf(ang);
    FhC[idx] = (_Float16)c;  FhS[idx] = (_Float16)s;  FhSn[idx] = (_Float16)(-s);
  }
  // Inverse H basis [u][kxsel]: e^{+i 2pi kxa u / 128}
  if (idx < HH * 64) {
    int u = idx >> 6, kxs = idx & 63;
    int kxa = (kxs < 32) ? kxs : kxs + 64;
    float ang = TWO_PI * (float)((kxa * u) & 127) * (1.0f / 128.0f);
    float c = cosf(ang), s = sinf(ang);
    IhC[idx] = (_Float16)c;  IhS[idx] = (_Float16)s;  IhSn[idx] = (_Float16)(-s);
  }
  // Inverse W basis (transposed [v][ky], v padded to 80): e^{+i 2pi ky v / 65} / (128*65)
  if (idx < WPAD * MM2) {
    int v = idx >> 5, ky = idx & 31;
    if (v < WOUT) {
      float scl = 1.0f / (128.0f * 65.0f);
      float ang = TWO_PI * (float)((ky * v) % 65) * (1.0f / 65.0f);
      IwCT[idx]  = (_Float16)(cosf(ang) * scl);
      IwSTn[idx] = (_Float16)(-sinf(ang) * scl);
    } else {
      IwCT[idx] = (_Float16)0.0f;
      IwSTn[idx] = (_Float16)0.0f;
    }
  }
}

// ---------------------------------------------------------------------------
// Kernel: repack weights fp32 [CI][CO][32][32] -> f16 [mode][o][i] (+ negated Wi)
// conj handled downstream: Yr = Xr*Wr + Xi*Wi ; Yi = Xi*Wr + Xr*(-Wi)
// ---------------------------------------------------------------------------
__global__ void k_repack_w(const float* __restrict__ w1r, const float* __restrict__ w1i,
                           const float* __restrict__ w2r, const float* __restrict__ w2i,
                           _Float16* __restrict__ WrT, _Float16* __restrict__ WiT,
                           _Float16* __restrict__ WinT) {
  const size_t idx = (size_t)blockIdx.x * 256 + threadIdx.x;  // < 2048*64*64
  const int m = (int)(idx >> 12);
  const int o = (int)((idx >> 6) & 63);
  const int i = (int)(idx & 63);
  const int kxs = m >> 5, ky = m & 31;
  const float* wr; const float* wi; int xx;
  if (kxs < 32) { wr = w1r; wi = w1i; xx = kxs; }
  else          { wr = w2r; wi = w2i; xx = kxs - 32; }
  const size_t s = (((size_t)i * NCO + o) * MM1 + xx) * MM2 + ky;
  float r = wr[s], im = wi[s];
  WrT[idx]  = (_Float16)r;
  WiT[idx]  = (_Float16)im;
  WinT[idx] = (_Float16)(-im);
}

// ---------------------------------------------------------------------------
// Stage 1: per (b, ci) image: 2D DFT restricted to 64x32 modes, all via WMMA.
//   A  = x(128x128) * Fw  -> Ar = x*cos, Ai = x*(-sin)   (K=128 over v)
//   X  = Fh(64x128) * A   -> Xr = C*Ar + S*Ai ; Xi = C*Ai + (-S)*Ar  (K=128 over u)
// Writes Xr/Xi f16 as [mode][b][ci] for stage-2 channel mixing.
// ---------------------------------------------------------------------------
__global__ __launch_bounds__(128) void k_stage1(
    const float* __restrict__ x,
    const _Float16* __restrict__ FwCT, const _Float16* __restrict__ FwSTn,
    const _Float16* __restrict__ FhC, const _Float16* __restrict__ FhS,
    const _Float16* __restrict__ FhSn,
    _Float16* __restrict__ Xr, _Float16* __restrict__ Xi) {
  alignas(16) __shared__ _Float16 sX[HH * WW];        // 32 KB, x image as f16 [u][v]
  alignas(16) __shared__ _Float16 sArT[MM2 * HH];     // 8 KB, Ar transposed [ky][u]
  alignas(16) __shared__ _Float16 sAiT[MM2 * HH];     // 8 KB

  const int blk = blockIdx.x;              // b*64 + ci
  const int b = blk >> 6, ci = blk & 63;
  const int t = threadIdx.x;
  const int wave = t >> 5, lane = t & 31;

  // ---- load + convert x image (coalesced float4) ----
  const float4* img4 = (const float4*)(x + (size_t)blk * (HH * WW));
#pragma unroll 4
  for (int j = t; j < HH * WW / 4; j += 128) {
    float4 f = img4[j];
    _Float16* d = sX + 4 * j;
    d[0] = (_Float16)f.x; d[1] = (_Float16)f.y;
    d[2] = (_Float16)f.z; d[3] = (_Float16)f.w;
  }
  __syncthreads();

  // ---- A-stage: each wave does 2 M-tiles of u (rows wave*32 .. +32) ----
  v8f accAr[2][2], accAi[2][2];
#pragma unroll
  for (int a = 0; a < 2; ++a)
#pragma unroll
    for (int n = 0; n < 2; ++n) { accAr[a][n] = zero8(); accAi[a][n] = zero8(); }

#pragma unroll
  for (int k0 = 0; k0 < WW; k0 += 32) {
    v16h a0 = load_frag_a(sX, (wave * 2 + 0) * 16, k0, WW);
    v16h a1 = load_frag_a(sX, (wave * 2 + 1) * 16, k0, WW);
#pragma unroll
    for (int nt = 0; nt < 2; ++nt) {
      v16h bc = load_frag_bt(FwCT,  nt * 16, k0, WW);
      v16h bs = load_frag_bt(FwSTn, nt * 16, k0, WW);
      accAr[0][nt] = wmma_f16(a0, bc, accAr[0][nt]);
      accAr[1][nt] = wmma_f16(a1, bc, accAr[1][nt]);
      accAi[0][nt] = wmma_f16(a0, bs, accAi[0][nt]);
      accAi[1][nt] = wmma_f16(a1, bs, accAi[1][nt]);
    }
  }
  // store A transposed into LDS: sArT[ky*128 + u]
  {
    const int nl = lane & 15, hi = lane >> 4;
#pragma unroll
    for (int mtl = 0; mtl < 2; ++mtl)
#pragma unroll
      for (int nt = 0; nt < 2; ++nt) {
        const int ub = (wave * 2 + mtl) * 16 + hi * 8;
        const int ky = nt * 16 + nl;
#pragma unroll
        for (int g = 0; g < 8; ++g) {
          sArT[ky * HH + ub + g] = (_Float16)accAr[mtl][nt][g];
          sAiT[ky * HH + ub + g] = (_Float16)accAi[mtl][nt][g];
        }
      }
  }
  __syncthreads();

  // ---- H-stage: each wave does one 16-row tile of selected kx ----
  v8f accXr[2], accXi[2];
  accXr[0] = zero8(); accXr[1] = zero8();
  accXi[0] = zero8(); accXi[1] = zero8();
#pragma unroll
  for (int k0 = 0; k0 < HH; k0 += 32) {
    v16h ac  = load_frag_a(FhC,  wave * 16, k0, HH);
    v16h as  = load_frag_a(FhS,  wave * 16, k0, HH);
    v16h asn = load_frag_a(FhSn, wave * 16, k0, HH);
#pragma unroll
    for (int nt = 0; nt < 2; ++nt) {
      v16h br = load_frag_bt(sArT, nt * 16, k0, HH);
      v16h bi = load_frag_bt(sAiT, nt * 16, k0, HH);
      accXr[nt] = wmma_f16(ac,  br, accXr[nt]);
      accXr[nt] = wmma_f16(as,  bi, accXr[nt]);
      accXi[nt] = wmma_f16(ac,  bi, accXi[nt]);
      accXi[nt] = wmma_f16(asn, br, accXi[nt]);
    }
  }
  // scatter store f16: Xr[(kx*32+ky)*2048 + b*64 + ci]
  {
    const int nl = lane & 15, hi = lane >> 4;
    const int boff = b * NCI + ci;
#pragma unroll
    for (int nt = 0; nt < 2; ++nt)
#pragma unroll
      for (int g = 0; g < 8; ++g) {
        const int kx = wave * 16 + hi * 8 + g;
        const int ky = nt * 16 + nl;
        const size_t idx = ((size_t)(kx * MM2 + ky)) * (BB * NCI) + boff;
        Xr[idx] = (_Float16)accXr[nt][g];
        Xi[idx] = (_Float16)accXi[nt][g];
      }
  }
}

// ---------------------------------------------------------------------------
// Stage 2: per mode m: complex channel mixing with conj weights.
//   Yr(32x64) = Xr*Wr + Xi*Wi ; Yi = Xi*Wr + Xr*(-Wi)   (K = 64 over ci)
// Writes Y f16 as [b][o][ky*64+kx] (i.e. transposed mode layout for stage 3).
// ---------------------------------------------------------------------------
__global__ __launch_bounds__(128) void k_stage2(
    const _Float16* __restrict__ Xr, const _Float16* __restrict__ Xi,
    const _Float16* __restrict__ WrT, const _Float16* __restrict__ WiT,
    const _Float16* __restrict__ WinT,
    _Float16* __restrict__ Yr, _Float16* __restrict__ Yi) {
  alignas(16) __shared__ _Float16 sXr[BB * NCI];     // 4 KB [b][i]
  alignas(16) __shared__ _Float16 sXi[BB * NCI];
  alignas(16) __shared__ _Float16 sWr[NCO * NCI];    // 8 KB [o][i] (BT layout)
  alignas(16) __shared__ _Float16 sWi[NCO * NCI];
  alignas(16) __shared__ _Float16 sWin[NCO * NCI];

  const int m = blockIdx.x;
  const int t = threadIdx.x;
  const int wave = t >> 5, lane = t & 31;

  {
    const unsigned* gXr = (const unsigned*)(Xr + (size_t)m * (BB * NCI));
    const unsigned* gXi = (const unsigned*)(Xi + (size_t)m * (BB * NCI));
    unsigned* dXr = (unsigned*)sXr; unsigned* dXi = (unsigned*)sXi;
#pragma unroll
    for (int j = t; j < BB * NCI / 2; j += 128) { dXr[j] = gXr[j]; dXi[j] = gXi[j]; }
    const unsigned* gWr  = (const unsigned*)(WrT  + (size_t)m * (NCO * NCI));
    const unsigned* gWi  = (const unsigned*)(WiT  + (size_t)m * (NCO * NCI));
    const unsigned* gWin = (const unsigned*)(WinT + (size_t)m * (NCO * NCI));
    unsigned* dWr = (unsigned*)sWr; unsigned* dWi = (unsigned*)sWi; unsigned* dWin = (unsigned*)sWin;
#pragma unroll
    for (int j = t; j < NCO * NCI / 2; j += 128) { dWr[j] = gWr[j]; dWi[j] = gWi[j]; dWin[j] = gWin[j]; }
  }
  __syncthreads();

  // wave handles output-channel tile o in [wave*16, wave*16+16)
  v8f aYr[2], aYi[2];
  aYr[0] = zero8(); aYr[1] = zero8(); aYi[0] = zero8(); aYi[1] = zero8();
#pragma unroll
  for (int k0 = 0; k0 < NCI; k0 += 32) {
    v16h xr0 = load_frag_a(sXr, 0,  k0, NCI);
    v16h xr1 = load_frag_a(sXr, 16, k0, NCI);
    v16h xi0 = load_frag_a(sXi, 0,  k0, NCI);
    v16h xi1 = load_frag_a(sXi, 16, k0, NCI);
    v16h bwr  = load_frag_bt(sWr,  wave * 16, k0, NCI);
    v16h bwi  = load_frag_bt(sWi,  wave * 16, k0, NCI);
    v16h bwin = load_frag_bt(sWin, wave * 16, k0, NCI);
    aYr[0] = wmma_f16(xr0, bwr,  aYr[0]);
    aYr[0] = wmma_f16(xi0, bwi,  aYr[0]);
    aYr[1] = wmma_f16(xr1, bwr,  aYr[1]);
    aYr[1] = wmma_f16(xi1, bwi,  aYr[1]);
    aYi[0] = wmma_f16(xi0, bwr,  aYi[0]);
    aYi[0] = wmma_f16(xr0, bwin, aYi[0]);
    aYi[1] = wmma_f16(xi1, bwr,  aYi[1]);
    aYi[1] = wmma_f16(xr1, bwin, aYi[1]);
  }
  // store: Y[b][o][ky*64+kx]
  {
    const int kx = m >> 5, ky = m & 31;
    const int mp = ky * 64 + kx;
    const int nl = lane & 15, hi = lane >> 4;
    const int o = wave * 16 + nl;
#pragma unroll
    for (int mtl = 0; mtl < 2; ++mtl)
#pragma unroll
      for (int g = 0; g < 8; ++g) {
        const int bb = mtl * 16 + hi * 8 + g;
        const size_t idx = ((size_t)(bb * NCO + o)) * NMODE + mp;
        Yr[idx] = (_Float16)aYr[mtl][g];
        Yi[idx] = (_Float16)aYi[mtl][g];
      }
  }
}

// ---------------------------------------------------------------------------
// Stage 3: per (b, o) image: inverse transform.
//   G(128x32) = Ih * Y : Gr = C*Yr + (-S)*Yi ; Gi = C*Yi + S*Yr   (K=64 over kx)
//   out(128x65) = Gr*IwC + Gi*(-IwS)                              (K=32 over ky)
// ---------------------------------------------------------------------------
__global__ __launch_bounds__(128) void k_stage3(
    const _Float16* __restrict__ Yr, const _Float16* __restrict__ Yi,
    const _Float16* __restrict__ IhC, const _Float16* __restrict__ IhS,
    const _Float16* __restrict__ IhSn,
    const _Float16* __restrict__ IwCT, const _Float16* __restrict__ IwSTn,
    float* __restrict__ out) {
  alignas(16) __shared__ _Float16 sYr[MM2 * 64];   // [ky][kx] — matches stage-2 layout
  alignas(16) __shared__ _Float16 sYi[MM2 * 64];
  alignas(16) __shared__ _Float16 sGr[HH * MM2];   // [u][ky]
  alignas(16) __shared__ _Float16 sGi[HH * MM2];

  const int bo = blockIdx.x;                // b*64 + o
  const int t = threadIdx.x;
  const int wave = t >> 5, lane = t & 31;

  {
    const unsigned* gYr = (const unsigned*)(Yr + (size_t)bo * NMODE);
    const unsigned* gYi = (const unsigned*)(Yi + (size_t)bo * NMODE);
    unsigned* dYr = (unsigned*)sYr; unsigned* dYi = (unsigned*)sYi;
#pragma unroll
    for (int j = t; j < NMODE / 2; j += 128) { dYr[j] = gYr[j]; dYi[j] = gYi[j]; }
  }
  __syncthreads();

  // ---- matmul 1: each wave handles u rows [wave*32, wave*32+32) ----
  v8f aGr[2][2], aGi[2][2];
#pragma unroll
  for (int a = 0; a < 2; ++a)
#pragma unroll
    for (int n = 0; n < 2; ++n) { aGr[a][n] = zero8(); aGi[a][n] = zero8(); }

#pragma unroll
  for (int k0 = 0; k0 < 64; k0 += 32) {
    v16h ac0  = load_frag_a(IhC,  wave * 32,      k0, 64);
    v16h ac1  = load_frag_a(IhC,  wave * 32 + 16, k0, 64);
    v16h as0  = load_frag_a(IhS,  wave * 32,      k0, 64);
    v16h as1  = load_frag_a(IhS,  wave * 32 + 16, k0, 64);
    v16h asn0 = load_frag_a(IhSn, wave * 32,      k0, 64);
    v16h asn1 = load_frag_a(IhSn, wave * 32 + 16, k0, 64);
#pragma unroll
    for (int nt = 0; nt < 2; ++nt) {
      v16h byr = load_frag_bt(sYr, nt * 16, k0, 64);
      v16h byi = load_frag_bt(sYi, nt * 16, k0, 64);
      aGr[0][nt] = wmma_f16(ac0,  byr, aGr[0][nt]);
      aGr[0][nt] = wmma_f16(asn0, byi, aGr[0][nt]);
      aGr[1][nt] = wmma_f16(ac1,  byr, aGr[1][nt]);
      aGr[1][nt] = wmma_f16(asn1, byi, aGr[1][nt]);
      aGi[0][nt] = wmma_f16(ac0,  byi, aGi[0][nt]);
      aGi[0][nt] = wmma_f16(as0,  byr, aGi[0][nt]);
      aGi[1][nt] = wmma_f16(ac1,  byi, aGi[1][nt]);
      aGi[1][nt] = wmma_f16(as1,  byr, aGi[1][nt]);
    }
  }
  // store G to LDS [u][ky]
  {
    const int nl = lane & 15, hi = lane >> 4;
#pragma unroll
    for (int mtl = 0; mtl < 2; ++mtl)
#pragma unroll
      for (int nt = 0; nt < 2; ++nt) {
        const int ub = wave * 32 + mtl * 16 + hi * 8;
        const int ky = nt * 16 + nl;
#pragma unroll
        for (int g = 0; g < 8; ++g) {
          sGr[(ub + g) * MM2 + ky] = (_Float16)aGr[mtl][nt][g];
          sGi[(ub + g) * MM2 + ky] = (_Float16)aGi[mtl][nt][g];
        }
      }
  }
  __syncthreads();

  // ---- matmul 2: out rows = same u rows; N = 80 padded (store only v<65) ----
  v16h agr0 = load_frag_a(sGr, wave * 32,      0, MM2);
  v16h agr1 = load_frag_a(sGr, wave * 32 + 16, 0, MM2);
  v16h agi0 = load_frag_a(sGi, wave * 32,      0, MM2);
  v16h agi1 = load_frag_a(sGi, wave * 32 + 16, 0, MM2);
  float* obase = out + (size_t)bo * (HH * WOUT);
  const int nl = lane & 15, hi = lane >> 4;
#pragma unroll
  for (int nt = 0; nt < 5; ++nt) {
    v16h bc = load_frag_bt(IwCT,  nt * 16, 0, MM2);
    v16h bs = load_frag_bt(IwSTn, nt * 16, 0, MM2);
    const int v = nt * 16 + nl;
#pragma unroll
    for (int mtl = 0; mtl < 2; ++mtl) {
      v8f acc = zero8();
      acc = wmma_f16(mtl ? agr1 : agr0, bc, acc);
      acc = wmma_f16(mtl ? agi1 : agi0, bs, acc);
      if (v < WOUT) {
#pragma unroll
        for (int g = 0; g < 8; ++g) {
          const int u = wave * 32 + mtl * 16 + hi * 8 + g;
          obase[u * WOUT + v] = acc[g];
        }
      }
    }
  }
}

// ---------------------------------------------------------------------------
// Host launcher
// ---------------------------------------------------------------------------
extern "C" void kernel_launch(void* const* d_in, const int* in_sizes, int n_in,
                              void* d_out, int out_size, void* d_ws, size_t ws_size,
                              hipStream_t stream) {
  (void)in_sizes; (void)n_in; (void)out_size; (void)ws_size;
  const float* x   = (const float*)d_in[0];
  const float* w1r = (const float*)d_in[1];
  const float* w1i = (const float*)d_in[2];
  const float* w2r = (const float*)d_in[3];
  const float* w2i = (const float*)d_in[4];
  float* out = (float*)d_out;

  // Workspace carve-out (f16 arrays, 256B aligned). Total ~80.5 MB.
  char* ws = (char*)d_ws;
  size_t off = 0;
  auto carve = [&](size_t elems) -> _Float16* {
    char* p = ws + off;
    off += elems * 2;
    off = (off + 255) & ~(size_t)255;
    return (_Float16*)p;
  };
  _Float16* FwCT  = carve(MM2 * WW);
  _Float16* FwSTn = carve(MM2 * WW);
  _Float16* FhC   = carve(64 * HH);
  _Float16* FhS   = carve(64 * HH);
  _Float16* FhSn  = carve(64 * HH);
  _Float16* IhC   = carve(HH * 64);
  _Float16* IhS   = carve(HH * 64);
  _Float16* IhSn  = carve(HH * 64);
  _Float16* IwCT  = carve(WPAD * MM2);
  _Float16* IwSTn = carve(WPAD * MM2);
  _Float16* WrT   = carve((size_t)NMODE * NCO * NCI);   // 16 MB
  _Float16* WiT   = carve((size_t)NMODE * NCO * NCI);
  _Float16* WinT  = carve((size_t)NMODE * NCO * NCI);
  _Float16* Xr    = carve((size_t)NMODE * BB * NCI);    // 8 MB
  _Float16* Xi    = carve((size_t)NMODE * BB * NCI);
  _Float16* Yrb   = carve((size_t)BB * NCO * NMODE);
  _Float16* Yib   = carve((size_t)BB * NCO * NMODE);

  k_init_basis<<<32, 256, 0, stream>>>(FwCT, FwSTn, FhC, FhS, FhSn,
                                       IhC, IhS, IhSn, IwCT, IwSTn);
  k_repack_w<<<(NMODE * NCO * NCI) / 256, 256, 0, stream>>>(w1r, w1i, w2r, w2i,
                                                            WrT, WiT, WinT);
  k_stage1<<<BB * NCI, 128, 0, stream>>>(x, FwCT, FwSTn, FhC, FhS, FhSn, Xr, Xi);
  k_stage2<<<NMODE, 128, 0, stream>>>(Xr, Xi, WrT, WiT, WinT, Yrb, Yib);
  k_stage3<<<BB * NCO, 128, 0, stream>>>(Yrb, Yib, IhC, IhS, IhSn, IwCT, IwSTn, out);
}
